// Attention__42322607735552
// MI455X (gfx1250) — matrix operand order repaired and checked by reading, not compiled, not run on hardware
//
#include <hip/hip_runtime.h>

#define N_DIM 1024
#define B_DIM 32
#define TQ    32
#define TK    128

typedef __attribute__((ext_vector_type(16))) __bf16 v16bf;
typedef __attribute__((ext_vector_type(8)))  __bf16 v8bf;
typedef __attribute__((ext_vector_type(2)))  __bf16 v2bf;
typedef __attribute__((ext_vector_type(8)))  float  v8f;
typedef int v4i_vec __attribute__((vector_size(16)));   // matches builtin param

__device__ __forceinline__ float fast_tanh(float x) {
  // tanh(x) = 1 - 2/(exp(2x)+1); exp via v_exp_f32
  float e = __builtin_amdgcn_exp2f(x * 2.8853900817779268f); // 2*log2(e)
  return 1.0f - 2.0f * __builtin_amdgcn_rcpf(e + 1.0f);
}
__device__ __forceinline__ float fast_exp(float x) {
  return __builtin_amdgcn_exp2f(x * 1.4426950408889634f);
}

// ---- async global -> LDS copy (16B per lane), with portable fallback -------
__device__ __forceinline__ void async_cp16(float* lds_dst, const float* gsrc) {
#if __has_builtin(__builtin_amdgcn_global_load_async_to_lds_b128)
  __builtin_amdgcn_global_load_async_to_lds_b128(
      (__attribute__((address_space(1))) v4i_vec*)gsrc,
      (__attribute__((address_space(3))) v4i_vec*)lds_dst, 0, 0);
#else
  *(float4*)lds_dst = *(const float4*)gsrc;
#endif
}
__device__ __forceinline__ void async_cp_wait() {
#if __has_builtin(__builtin_amdgcn_s_wait_asynccnt)
  __builtin_amdgcn_s_wait_asynccnt(0);
#endif
}

// ---------------------------------------------------------------------------
// bf16 conversion passes.
// X (T,B,N) f32 -> Xb (B*T, N) bf16 row-major (GEMM A operand)
// ---------------------------------------------------------------------------
__global__ __launch_bounds__(256)
void cvt_x_kernel(const float* __restrict__ X, __bf16* __restrict__ Xb, int T)
{
  size_t e = ((size_t)blockIdx.x * 256 + threadIdx.x) * 2;  // element pair
  int    n = (int)(e & (N_DIM - 1));
  int    m = (int)(e >> 10);
  int    b = m / T;
  int    t = m - b * T;
  const float* src = X + (size_t)t * (B_DIM * N_DIM) + (size_t)b * N_DIM + n;
  v2bf o; o[0] = (__bf16)src[0]; o[1] = (__bf16)src[1];
  *(v2bf*)(Xb + e) = o;
}

__global__ __launch_bounds__(256)
void cvt_w_kernel(const float* __restrict__ W, __bf16* __restrict__ Wb)
{
  size_t e = ((size_t)blockIdx.x * 256 + threadIdx.x) * 2;
  v2bf o; o[0] = (__bf16)W[e]; o[1] = (__bf16)W[e + 1];
  *(v2bf*)(Wb + e) = o;
}

// ---------------------------------------------------------------------------
// Projection GEMM (bf16 operands, f32 accumulate):
//   Y[m, n] = sum_e A[m, e] * W[n, e]
// One wave32 computes a 32x64 block: 2 A-frags x 4 B-frags -> 8 WMMAs per
// k-step, software-pipelined (next step's 12 loads overlap current WMMAs).
// ---------------------------------------------------------------------------
struct KFrags {
  v16bf a0, a1;           // A rows [base..base+15], [base+16..base+31]
  v16bf b0, b1, b2, b3;   // B cols [nbase+u*16 .. +15]
};

__device__ __forceinline__ v16bf load_afrag(const __bf16* arow, int ka) {
  v8bf lo = *(const v8bf*)(arow + ka);
  v8bf hi = *(const v8bf*)(arow + ka + 16);
  v16bf af;
#pragma unroll
  for (int j = 0; j < 8; ++j) { af[j] = lo[j]; af[j + 8] = hi[j]; }
  return af;
}

__device__ __forceinline__ KFrags load_kstep(const __bf16* arow0,
                                             const __bf16* arow1,
                                             const __bf16* wrow, int kk,
                                             int half) {
  KFrags f;
  int ka = kk + half * 8;    // A layout: lanes 0-15 K=+0..7/+16..23; 16-31 +8/+24
  int kb = kk + half * 8;    // B layout: lanes 0-15 K=+0..7/+16..23; 16-31 +8/+24
  f.a0 = load_afrag(arow0, ka);
  f.a1 = load_afrag(arow1, ka);
  f.b0 = load_afrag(wrow + (size_t)0 * 16 * N_DIM, kb);
  f.b1 = load_afrag(wrow + (size_t)1 * 16 * N_DIM, kb);
  f.b2 = load_afrag(wrow + (size_t)2 * 16 * N_DIM, kb);
  f.b3 = load_afrag(wrow + (size_t)3 * 16 * N_DIM, kb);
  return f;
}

#define WMMA_BF16(ACC, AF, BF)                                                \
  ACC = __builtin_amdgcn_wmma_f32_16x16x32_bf16(false, AF, false, BF,         \
                                                (short)0, ACC, false, false)

__global__ __launch_bounds__(128)
void proj_wmma_kernel(const __bf16* __restrict__ A,   // (M, N) bf16
                      const __bf16* __restrict__ W,   // (N, N) bf16
                      float* __restrict__ Y,          // (M, N) f32
                      int M)
{
  const int N = N_DIM;
  const int nstrips = N / 64;                          // 16
  int wave  = (blockIdx.x * blockDim.x + threadIdx.x) >> 5;
  int lane  = threadIdx.x & 31;
  int mblk  = wave / nstrips;                          // 32-row block index
  int nstr  = wave % nstrips;                          // 64-col strip index
  if (mblk * 32 >= M) return;

  int half = lane >> 4;          // 0: lanes 0-15, 1: lanes 16-31
  int l16  = lane & 15;

  const __bf16* arow0 = A + (size_t)(mblk * 32 + l16) * N;
  const __bf16* arow1 = arow0 + (size_t)16 * N;
  const __bf16* wrow  = W + (size_t)(nstr * 64 + l16) * N;

  v8f c00 = {}, c01 = {}, c02 = {}, c03 = {};
  v8f c10 = {}, c11 = {}, c12 = {}, c13 = {};

  KFrags cur = load_kstep(arow0, arow1, wrow, 0, half);
  for (int kk = 32; kk < N; kk += 32) {
    KFrags nxt = load_kstep(arow0, arow1, wrow, kk, half);
    WMMA_BF16(c00, cur.a0, cur.b0);
    WMMA_BF16(c01, cur.a0, cur.b1);
    WMMA_BF16(c02, cur.a0, cur.b2);
    WMMA_BF16(c03, cur.a0, cur.b3);
    WMMA_BF16(c10, cur.a1, cur.b0);
    WMMA_BF16(c11, cur.a1, cur.b1);
    WMMA_BF16(c12, cur.a1, cur.b2);
    WMMA_BF16(c13, cur.a1, cur.b3);
    cur = nxt;
  }
  WMMA_BF16(c00, cur.a0, cur.b0);
  WMMA_BF16(c01, cur.a0, cur.b1);
  WMMA_BF16(c02, cur.a0, cur.b2);
  WMMA_BF16(c03, cur.a0, cur.b3);
  WMMA_BF16(c10, cur.a1, cur.b0);
  WMMA_BF16(c11, cur.a1, cur.b1);
  WMMA_BF16(c12, cur.a1, cur.b2);
  WMMA_BF16(c13, cur.a1, cur.b3);

  // C/D layout: VGPR i -> (row = rowbase + i + half*8, col = colbase + l16)
  float* base0 = Y + (size_t)(mblk * 32) * N + (size_t)(nstr * 64 + l16);
  float* base1 = base0 + (size_t)16 * N;
#pragma unroll
  for (int i = 0; i < 8; ++i) {
    size_t r = (size_t)(i + half * 8) * N;
    base0[r + 0]  = c00[i];
    base0[r + 16] = c01[i];
    base0[r + 32] = c02[i];
    base0[r + 48] = c03[i];
    base1[r + 0]  = c10[i];
    base1[r + 16] = c11[i];
    base1[r + 32] = c12[i];
    base1[r + 48] = c13[i];
  }
}

// ---------------------------------------------------------------------------
// Normalize the energy vector: v = linear_att / ||linear_att|| * scalar
// ---------------------------------------------------------------------------
__global__ __launch_bounds__(256)
void norm_v_kernel(const float* __restrict__ la,
                   const float* __restrict__ scalar,
                   float* __restrict__ v_out)
{
  __shared__ float red[256];
  int tid = threadIdx.x;
  float s = 0.f;
  for (int i = tid; i < N_DIM; i += 256) { float x = la[i]; s += x * x; }
  red[tid] = s;
  __syncthreads();
  for (int off = 128; off > 0; off >>= 1) {
    if (tid < off) red[tid] += red[tid + off];
    __syncthreads();
  }
  float scale = scalar[0] * rsqrtf(red[0]);
  for (int i = tid; i < N_DIM; i += 256) v_out[i] = la[i] * scale;
}

// ---------------------------------------------------------------------------
// Scores + softmax + context. 8 waves/block, all sharing one batch b:
// pk and keys rows are staged into LDS in 8-row chunks (async copies) and
// consumed by all 8 waves, cutting L2 traffic 8x.
// ---------------------------------------------------------------------------
#define ROWS    8
#define LSTRIDE (N_DIM + 16)   // float stride, keeps 16B alignment, depads banks

__global__ __launch_bounds__(256)
void attn_kernel(const float* __restrict__ pq,     // (B*TQ, N)
                 const float* __restrict__ pk,     // (B*TK, N)
                 const float* __restrict__ keys,   // (TK, B, N)
                 const float* __restrict__ vvec,   // (N)
                 const float* __restrict__ bias,   // (N)
                 float* __restrict__ out)          // full output buffer
{
  const int N = N_DIM;
  __shared__ float rowbuf[ROWS * LSTRIDE];   // 33,280 B
  __shared__ float sm[8][TK];                // 4 KB

  int tid  = threadIdx.x;
  int warp = tid >> 5;
  int lane = tid & 31;
  int bq0  = blockIdx.x * 8;          // 8 consecutive (b,q): same b
  int b    = bq0 >> 5;
  int q    = (bq0 & 31) + warp;

  const float* pqrow = pq + (size_t)(b * TQ + q) * N;
  const float* pkb   = pk + (size_t)(b * TK) * N;

  // Per-lane n-slice: n = i*128 + lane*4 + {0..3}
  float4 pqv[8], vv[8];
#pragma unroll
  for (int i = 0; i < 8; ++i) {
    int n = i * 128 + lane * 4;
    float4 p  = *(const float4*)(pqrow + n);
    float4 bb = *(const float4*)(bias + n);
    pqv[i].x = p.x + bb.x; pqv[i].y = p.y + bb.y;
    pqv[i].z = p.z + bb.z; pqv[i].w = p.w + bb.w;
    vv[i] = *(const float4*)(vvec + n);
  }

  // ---- scores: 16 chunks of 8 pk rows staged in LDS ----
  for (int c = 0; c < TK / ROWS; ++c) {
    __syncthreads();                       // prior chunk fully consumed
#pragma unroll
    for (int j = 0; j < 8; ++j) {          // 2048 float4 over 256 threads
      int idx = tid + j * 256;
      int r   = idx >> 8;                  // row within chunk
      int c4  = idx & 255;                 // float4 column
      async_cp16(&rowbuf[r * LSTRIDE + c4 * 4],
                 pkb + (size_t)(c * ROWS + r) * N + c4 * 4);
    }
    async_cp_wait();
    __syncthreads();
#pragma unroll 1
    for (int r = 0; r < ROWS; ++r) {
      const float* row = &rowbuf[r * LSTRIDE];
      float s = 0.f;
#pragma unroll
      for (int i = 0; i < 8; ++i) {
        float4 pkv = *(const float4*)(row + i * 128 + lane * 4);
        s += fast_tanh(pqv[i].x + pkv.x) * vv[i].x;
        s += fast_tanh(pqv[i].y + pkv.y) * vv[i].y;
        s += fast_tanh(pqv[i].z + pkv.z) * vv[i].z;
        s += fast_tanh(pqv[i].w + pkv.w) * vv[i].w;
      }
#pragma unroll
      for (int off = 16; off > 0; off >>= 1) s += __shfl_xor(s, off, 32);
      if (lane == 0) sm[warp][c * ROWS + r] = s;
    }
  }
  __syncthreads();

  // ---- softmax over TK=128: each lane owns 4 scores ----
  float s0 = sm[warp][lane];
  float s1 = sm[warp][lane + 32];
  float s2 = sm[warp][lane + 64];
  float s3 = sm[warp][lane + 96];
  float mx = fmaxf(fmaxf(s0, s1), fmaxf(s2, s3));
#pragma unroll
  for (int off = 16; off > 0; off >>= 1) mx = fmaxf(mx, __shfl_xor(mx, off, 32));
  float e0 = fast_exp(s0 - mx), e1 = fast_exp(s1 - mx);
  float e2 = fast_exp(s2 - mx), e3 = fast_exp(s3 - mx);
  float sum = e0 + e1 + e2 + e3;
#pragma unroll
  for (int off = 16; off > 0; off >>= 1) sum += __shfl_xor(sum, off, 32);
  float inv = __builtin_amdgcn_rcpf(sum);
  sm[warp][lane]      = e0 * inv;
  sm[warp][lane + 32] = e1 * inv;
  sm[warp][lane + 64] = e2 * inv;
  sm[warp][lane + 96] = e3 * inv;
  __syncthreads();

  // ---- context[n] = sum_k attn[k] * keys[k, b, n], keys chunks via LDS ----
  float4 acc[8];
#pragma unroll
  for (int i = 0; i < 8; ++i) { acc[i].x = acc[i].y = acc[i].z = acc[i].w = 0.f; }

  for (int c = 0; c < TK / ROWS; ++c) {
    __syncthreads();
#pragma unroll
    for (int j = 0; j < 8; ++j) {
      int idx = tid + j * 256;
      int r   = idx >> 8;
      int c4  = idx & 255;
      async_cp16(&rowbuf[r * LSTRIDE + c4 * 4],
                 keys + (size_t)(c * ROWS + r) * (B_DIM * N) +
                     (size_t)b * N + c4 * 4);
    }
    async_cp_wait();
    __syncthreads();
#pragma unroll 1
    for (int r = 0; r < ROWS; ++r) {
      float a = sm[warp][c * ROWS + r];
      const float* row = &rowbuf[r * LSTRIDE];
#pragma unroll
      for (int i = 0; i < 8; ++i) {
        float4 kv = *(const float4*)(row + i * 128 + lane * 4);
        acc[i].x += a * kv.x; acc[i].y += a * kv.y;
        acc[i].z += a * kv.z; acc[i].w += a * kv.w;
      }
    }
  }

  const size_t ctx_off    = (size_t)TQ * B_DIM * N;       // output #1 start
  const size_t concat_off = 2 * (size_t)TQ * B_DIM * N;   // output #2 start
#pragma unroll
  for (int i = 0; i < 8; ++i) {
    int n = i * 128 + lane * 4;
    *(float4*)(out + ctx_off + (size_t)q * (B_DIM * N) + (size_t)b * N + n) =
        acc[i];
    *(float4*)(out + concat_off + (size_t)q * (B_DIM * 2 * N) +
               (size_t)b * (2 * N) + N + n) = acc[i];
  }
}

// ---------------------------------------------------------------------------
// Copy query into output #0 and concat head (cols 0..N-1)
// ---------------------------------------------------------------------------
__global__ __launch_bounds__(256)
void copy_query_kernel(const float* __restrict__ query, float* __restrict__ out)
{
  const int N = N_DIM;
  size_t idx = (size_t)blockIdx.x * blockDim.x + threadIdx.x; // < TQ*B*N
  float x = query[idx];
  out[idx] = x;
  size_t t = idx / (B_DIM * N);
  size_t r = idx - t * (B_DIM * N);
  size_t b = r / N;
  size_t n = r - b * N;
  out[2 * (size_t)TQ * B_DIM * N + t * (B_DIM * 2 * N) + b * (2 * N) + n] = x;
}

// ---------------------------------------------------------------------------
extern "C" void kernel_launch(void* const* d_in, const int* in_sizes, int n_in,
                              void* d_out, int out_size, void* d_ws, size_t ws_size,
                              hipStream_t stream) {
  (void)in_sizes; (void)n_in; (void)out_size; (void)ws_size;
  const float* query = (const float*)d_in[0];   // (TQ, B, N)
  const float* keys  = (const float*)d_in[1];   // (TK, B, N)
  const float* Wq    = (const float*)d_in[2];   // (N, N)
  const float* Wk    = (const float*)d_in[3];   // (N, N)
  const float* la    = (const float*)d_in[4];   // (N)
  const float* nsc   = (const float*)d_in[5];   // (1)
  const float* nbias = (const float*)d_in[6];   // (N)
  float* out = (float*)d_out;

  // workspace layout
  char* ws = (char*)d_ws;
  float*  pq  = (float*)ws;                               ws += (size_t)4  << 20; // (B*TQ,N) f32
  float*  pk  = (float*)ws;                               ws += (size_t)16 << 20; // (B*TK,N) f32
  float*  vw  = (float*)ws;                               ws += 4096;             // (N) f32
  __bf16* qb  = (__bf16*)ws;                              ws += (size_t)2  << 20; // (B*TQ,N) bf16
  __bf16* kb  = (__bf16*)ws;                              ws += (size_t)8  << 20; // (B*TK,N) bf16
  __bf16* wqb = (__bf16*)ws;                              ws += (size_t)2  << 20; // (N,N) bf16
  __bf16* wkb = (__bf16*)ws;                              ws += (size_t)2  << 20; // (N,N) bf16

  // bf16 conversion passes (2 elements/thread)
  cvt_x_kernel<<<2048, 256, 0, stream>>>(query, qb, TQ);   // 1M elems
  cvt_x_kernel<<<8192, 256, 0, stream>>>(keys,  kb, TK);   // 4M elems
  cvt_w_kernel<<<2048, 256, 0, stream>>>(Wq, wqb);         // 1M elems
  cvt_w_kernel<<<2048, 256, 0, stream>>>(Wk, wkb);         // 1M elems

  // v = linear_att / ||linear_att|| * normalize_scalar
  norm_v_kernel<<<1, 256, 0, stream>>>(la, nsc, vw);

  // pq: 32 mblks * 16 strips = 512 waves -> 128 blocks of 4 waves
  proj_wmma_kernel<<<128, 128, 0, stream>>>(qb, wqb, pq, B_DIM * TQ);
  // pk: 128 mblks * 16 strips = 2048 waves -> 512 blocks
  proj_wmma_kernel<<<512, 128, 0, stream>>>(kb, wkb, pk, B_DIM * TK);

  // scores + softmax + context: 1024 (b,q) waves, 8 per block (shared b)
  attn_kernel<<<128, 256, 0, stream>>>(pq, pk, keys, vw, nbias, out);

  // query passthrough outputs: TQ*B*N / 256 = 4096 blocks
  copy_query_kernel<<<4096, 256, 0, stream>>>(query, out);
}
